// HGNNScheduler_63840393888127
// MI455X (gfx1250) — matrix operand order repaired and checked by reading, not compiled, not run on hardware
//
#include <hip/hip_runtime.h>
#include <hip/hip_bf16.h>
#include <math.h>

typedef __attribute__((ext_vector_type(16))) _Float16 v16h;
typedef __attribute__((ext_vector_type(8)))  _Float16 v8h;
typedef __attribute__((ext_vector_type(8)))  float    v8f;

#define B_   64
#define N_   4096
#define M_   128
#define DOPE 6
#define DMA  3
#define DOUT 8
#define NC   128           // N-chunks for pass-1 partial softmax
#define CHUNK (N_ / NC)    // 32 rows per chunk
#define FH_STRIDE 40       // padded LDS row stride (halves): 80B -> conflict-free b128 reads
#define FH_TILE (16 * FH_STRIDE)   // 640 halves per wave tile

__device__ __forceinline__ float leaky(float x) { return x >= 0.f ? x : 0.2f * x; }
__device__ __forceinline__ float sigm(float x)  { return 1.f / (1.f + __expf(-x)); }

// ---------------------------------------------------------------------------
// Kernel 1: feat_src = ope_feat @ W_src (stored f16 for the WMMA B fragment),
//           el = feat_src @ attn_l (f32)
// ---------------------------------------------------------------------------
__global__ void k_src(const float* __restrict__ ope, const float* __restrict__ Ws,
                      const float* __restrict__ al,
                      _Float16* __restrict__ fsh, float* __restrict__ elw)
{
    int r = blockIdx.x * blockDim.x + threadIdx.x;     // row in [0, B*N)
    if (r >= B_ * N_) return;
    float x[DOPE];
#pragma unroll
    for (int j = 0; j < DOPE; ++j) x[j] = ope[(size_t)r * DOPE + j];
    float el = 0.f;
    v8h row;
#pragma unroll
    for (int d = 0; d < DOUT; ++d) {
        float acc = 0.f;
#pragma unroll
        for (int j = 0; j < DOPE; ++j) acc = fmaf(x[j], Ws[j * DOUT + d], acc);
        row[d] = (_Float16)acc;
        el = fmaf(acc, al[d], el);
    }
    *(v8h*)(fsh + (size_t)r * DOUT) = row;             // 16B store
    elw[r] = el;
}

// ---------------------------------------------------------------------------
// Kernel 2: feat_dst = ma_feat @ W_dst, er = feat_dst @ attn_r, ekk = lrelu(2*er)
// ---------------------------------------------------------------------------
__global__ void k_dst(const float* __restrict__ ma, const float* __restrict__ Wd,
                      const float* __restrict__ ar,
                      float* __restrict__ fdw, float* __restrict__ erw,
                      float* __restrict__ ekkw)
{
    int r = blockIdx.x * blockDim.x + threadIdx.x;     // row in [0, B*M)
    if (r >= B_ * M_) return;
    float x[DMA];
#pragma unroll
    for (int j = 0; j < DMA; ++j) x[j] = ma[(size_t)r * DMA + j];
    float er = 0.f;
#pragma unroll
    for (int d = 0; d < DOUT; ++d) {
        float acc = 0.f;
#pragma unroll
        for (int j = 0; j < DMA; ++j) acc = fmaf(x[j], Wd[j * DOUT + d], acc);
        fdw[(size_t)r * DOUT + d] = acc;
        er = fmaf(acc, ar[d], er);
    }
    erw[r]  = er;
    ekkw[r] = leaky(2.f * er);
}

// ---------------------------------------------------------------------------
// Kernel 3: streaming pass 1 — per (b, chunk) online (max, sum-exp) partials.
// One wave per block; lane owns 4 m-columns -> b128 loads of adj and proc.
// ---------------------------------------------------------------------------
__global__ void __launch_bounds__(32)
k_pass1(const int* __restrict__ adj, const int* __restrict__ bidx,
        const float* __restrict__ proc, const float* __restrict__ elw,
        const float* __restrict__ erw, const float* __restrict__ we,
        const float* __restrict__ ae,
        float* __restrict__ pmax, float* __restrict__ psum)
{
    int l  = threadIdx.x;                 // lane, 0..31
    int b  = blockIdx.x / NC;
    int c  = blockIdx.x % NC;
    int b2 = bidx[b];

    float wea = 0.f;
#pragma unroll
    for (int d = 0; d < DOUT; ++d) wea = fmaf(we[d], ae[d], wea);

    int m0 = l * 4;
    float4 er4 = *(const float4*)&erw[(size_t)b * M_ + m0];
    float er_[4] = { er4.x, er4.y, er4.z, er4.w };

    int n0 = c * CHUNK;
    const int4*   ap  = (const int4*)  (adj  + ((size_t)b2 * N_ + n0) * M_) + l;
    const float4* pp  = (const float4*)(proc + ((size_t)b  * N_ + n0) * M_) + l;
    const float*  elp = elw + (size_t)b * N_ + n0;

    float mx_[4] = { -INFINITY, -INFINITY, -INFINITY, -INFINITY };
    float s_[4]  = { 0.f, 0.f, 0.f, 0.f };

    for (int i = 0; i < CHUNK; ++i) {
        __builtin_prefetch(&ap[(size_t)(i + 4) * (M_ / 4)], 0, 0);   // global_prefetch_b8
        __builtin_prefetch(&pp[(size_t)(i + 4) * (M_ / 4)], 0, 0);
        int4   a4 = ap[(size_t)i * (M_ / 4)];
        float4 p4 = pp[(size_t)i * (M_ / 4)];
        float  en = elp[i];
        int   a_[4] = { a4.x, a4.y, a4.z, a4.w };
        float p_[4] = { p4.x, p4.y, p4.z, p4.w };
#pragma unroll
        for (int j = 0; j < 4; ++j) {
            if (a_[j]) {
                float e  = leaky(en + er_[j] + p_[j] * wea);
                float nm = fmaxf(mx_[j], e);
                s_[j]  = s_[j] * __expf(mx_[j] - nm) + __expf(e - nm);
                mx_[j] = nm;
            }
        }
    }
    size_t o = ((size_t)b * NC + c) * M_ + m0;
    *(float4*)&pmax[o] = make_float4(mx_[0], mx_[1], mx_[2], mx_[3]);
    *(float4*)&psum[o] = make_float4(s_[0],  s_[1],  s_[2],  s_[3]);
}

// ---------------------------------------------------------------------------
// Kernel 4: merge chunk partials + ekk row -> global max, 1/denominator, alpha_kk
// ---------------------------------------------------------------------------
__global__ void k_combine(const float* __restrict__ pmax, const float* __restrict__ psum,
                          const float* __restrict__ ekkw,
                          float* __restrict__ MxA, float* __restrict__ rinvA,
                          float* __restrict__ akkA)
{
    int t = blockIdx.x * blockDim.x + threadIdx.x;     // [0, B*M)
    if (t >= B_ * M_) return;
    int b = t / M_, m = t % M_;
    float ekk = ekkw[t];
    float mx = ekk;
    for (int c = 0; c < NC; ++c)
        mx = fmaxf(mx, pmax[((size_t)b * NC + c) * M_ + m]);
    float S = __expf(ekk - mx);
    for (int c = 0; c < NC; ++c) {
        size_t o = ((size_t)b * NC + c) * M_ + m;
        S += psum[o] * __expf(pmax[o] - mx);           // mx==-inf partials contribute 0
    }
    float rinv = 1.f / S;
    MxA[t]   = mx;
    rinvA[t] = rinv;
    akkA[t]  = __expf(ekk - mx) * rinv;
}

// ---------------------------------------------------------------------------
// Kernel 5: streaming pass 2 — alpha-weighted sums via WMMA.
// Block = (b, 16-m tile); 8 waves, wave w owns n in [w*512, w*512+512).
// Per 32-row K step:
//   - stage el row-block (32 f32) and the feat tile TRANSPOSED into LDS
//     ([16 d][32 k], row stride 40 halves = 80B -> conflict-free b128 reads;
//     d=8..15 pre-zeroed once),
//   - B fragment = 2x ds_load_b128 per lane (branchless),
//   - el broadcast = 4x ds_load_b128 (float4),
//   - A fragment = alpha (f16) per ISA 16-bit 16x32 A layout; adj/proc loads
//     batched up-front so they clause; select (not branch) masks adj==0,
//   - v_wmma_f32_16x16x32_f16 accumulate; wp = sum(alpha*proc) rides in VALU.
// Cross-wave reduce via LDS ds_add_f32, then sigmoid epilogue.
// ---------------------------------------------------------------------------
__global__ void __launch_bounds__(256)
k_pass2(const int* __restrict__ adj, const int* __restrict__ bidx,
        const float* __restrict__ proc, const float* __restrict__ elw,
        const _Float16* __restrict__ fsh, const float* __restrict__ fdw,
        const float* __restrict__ erw,
        const float* __restrict__ MxA, const float* __restrict__ rinvA,
        const float* __restrict__ akkA,
        const float* __restrict__ we, const float* __restrict__ ae,
        float* __restrict__ out)
{
    __shared__ __align__(16) float    els[8][32];
    __shared__ __align__(16) _Float16 fh[8 * FH_TILE];
    __shared__ float bsum[16][DOUT];
    __shared__ float wpS[16];

    int tid = threadIdx.x;
    int w   = tid >> 5;          // wave id (wave32)
    int l   = tid & 31;          // lane id
    int b   = blockIdx.x >> 3;
    int mt  = blockIdx.x & 7;
    int b2  = bidx[b];

    if (tid < 16 * DOUT) bsum[tid >> 3][tid & 7] = 0.f;
    if (tid < 16)        wpS[tid] = 0.f;

    // zero this wave's LDS feat tile once (covers the d=8..15 padding columns)
    _Float16* fhw = fh + w * FH_TILE;
    {
        unsigned int* z = (unsigned int*)fhw;
#pragma unroll
        for (int i = 0; i < FH_TILE / 2; i += 32) z[i + l] = 0u;
    }
    __syncthreads();

    float wea = 0.f;
#pragma unroll
    for (int d = 0; d < DOUT; ++d) wea = fmaf(we[d], ae[d], wea);

    int   mloc = l & 15;
    int   mg   = mt * 16 + mloc;                 // this lane's A-matrix row (m index)
    float Mx   = MxA  [(size_t)b * M_ + mg];
    float rin  = rinvA[(size_t)b * M_ + mg];
    float erv  = erw  [(size_t)b * M_ + mg];

    const int*   ap2 = adj  + (size_t)b2 * N_ * M_ + mg;
    const float* pp2 = proc + (size_t)b  * N_ * M_ + mg;

    int col   = l & 15;
    int kb    = (l < 16) ? 0 : 16;   // unused; B fragment also indexes K via kbase
    int kbase = (l < 16) ? 0 : 8;    // A-fragment K base (ISA 16-bit A 16x32 layout)

    v8f   cf = {};
    float wp = 0.f;

    const int nbeg = w * 512, nend = nbeg + 512;
    for (int n0 = nbeg; n0 < nend; n0 += 32) {
        // ---- stage tiles (per-wave LDS region; same-wave DS ops retire in order)
        els[w][l] = elw[(size_t)b * N_ + n0 + l];
        v8h frow = ((const v8h*)(fsh + ((size_t)b * N_ + n0) * DOUT))[l];  // global b128
#pragma unroll
        for (int d = 0; d < DOUT; ++d)
            fhw[d * FH_STRIDE + l] = frow[d];                              // transpose
        asm volatile("" ::: "memory");

        // ---- B fragment: lane = column (l&15); K rows kbase..kbase+7, kbase+16..kbase+23 -> 2x ds_load_b128
        v8h blo = *(const v8h*)(fhw + col * FH_STRIDE + kbase);
        v8h bhi = *(const v8h*)(fhw + col * FH_STRIDE + kbase + 16);
        v16h bf;
#pragma unroll
        for (int e = 0; e < 8; ++e) { bf[e] = blo[e]; bf[e + 8] = bhi[e]; }

        // ---- el broadcast: 4x ds_load_b128 (same-address broadcast per 16-lane group)
        float4 e0 = *(const float4*)&els[w][kbase];
        float4 e1 = *(const float4*)&els[w][kbase + 4];
        float4 e2 = *(const float4*)&els[w][kbase + 16];
        float4 e3 = *(const float4*)&els[w][kbase + 20];
        float elv[16] = { e0.x, e0.y, e0.z, e0.w, e1.x, e1.y, e1.z, e1.w,
                          e2.x, e2.y, e2.z, e2.w, e3.x, e3.y, e3.z, e3.w };

        // ---- batched streaming loads for the A fragment
        int   av[16];
        float pv[16];
#pragma unroll
        for (int e = 0; e < 16; ++e) {
            int k = kbase + (e < 8 ? e : e + 8);
            av[e] = ap2[(size_t)(n0 + k) * M_];
            pv[e] = pp2[(size_t)(n0 + k) * M_];
        }
        __builtin_prefetch(&ap2[(size_t)(n0 + 32 + mloc) * M_], 0, 0);
        __builtin_prefetch(&pp2[(size_t)(n0 + 32 + mloc) * M_], 0, 0);

        // ---- A fragment: alpha in f16 (branchless select masks adj==0)
        v16h af;
#pragma unroll
        for (int e = 0; e < 16; ++e) {
            float ev = leaky(elv[e] + erv + pv[e] * wea);
            float a  = av[e] ? (__expf(ev - Mx) * rin) : 0.f;   // alpha in [0,1]
            wp = fmaf(a, pv[e], wp);
            af[e] = (_Float16)a;
        }

        cf = __builtin_amdgcn_wmma_f32_16x16x32_f16(false, af, false, bf,
                                                    (short)0, cf, false, false);
    }

    // ---- reduce C fragments across waves:
    // lane c<16 -> (M=r, N=c); lane c>=16 -> (M=8+r, N=c-16)
    if (col < DOUT) {
#pragma unroll
        for (int r = 0; r < 8; ++r) {
            int mrow = r + ((l >> 4) << 3);
            atomicAdd(&bsum[mrow][col], cf[r]);
        }
    }
    atomicAdd(&wpS[mloc], wp);
    __syncthreads();

    // ---- epilogue: out(b,m,d) = sigmoid(wp[m]*w_edge[d] + bsrc[m][d] + feat_dst*alpha_kk)
    if (tid < 16 * DOUT) {
        int mr = tid >> 3, d = tid & 7;
        int mgg = mt * 16 + mr;
        size_t gm = (size_t)b * M_ + mgg;
        float v = wpS[mr] * we[d] + bsum[mr][d] + fdw[gm * DOUT + d] * akkA[gm];
        out[gm * DOUT + d] = sigm(v);
    }
}

// ---------------------------------------------------------------------------
extern "C" void kernel_launch(void* const* d_in, const int* in_sizes, int n_in,
                              void* d_out, int out_size, void* d_ws, size_t ws_size,
                              hipStream_t stream)
{
    const int*   adj  = (const int*)  d_in[0];
    const int*   bidx = (const int*)  d_in[1];
    const float* ope  = (const float*)d_in[2];
    const float* ma   = (const float*)d_in[3];
    const float* proc = (const float*)d_in[4];
    const float* Ws   = (const float*)d_in[5];
    const float* Wd   = (const float*)d_in[6];
    const float* we   = (const float*)d_in[7];
    const float* al   = (const float*)d_in[8];
    const float* ar   = (const float*)d_in[9];
    const float* ae   = (const float*)d_in[10];
    float* out = (float*)d_out;

    // workspace carve-out (256B aligned slices, ~14 MB total)
    char* p = (char*)d_ws;
    auto carve = [&](size_t bytes) -> char* {
        char* q = p;
        p += (bytes + 255) & ~(size_t)255;
        return q;
    };
    _Float16* fsh  = (_Float16*)carve((size_t)B_ * N_ * DOUT * sizeof(_Float16));
    float*    elw  = (float*)   carve((size_t)B_ * N_ * sizeof(float));
    float*    fdw  = (float*)   carve((size_t)B_ * M_ * DOUT * sizeof(float));
    float*    erw  = (float*)   carve((size_t)B_ * M_ * sizeof(float));
    float*    ekkw = (float*)   carve((size_t)B_ * M_ * sizeof(float));
    float*    pmax = (float*)   carve((size_t)B_ * NC * M_ * sizeof(float));
    float*    psum = (float*)   carve((size_t)B_ * NC * M_ * sizeof(float));
    float*    MxA  = (float*)   carve((size_t)B_ * M_ * sizeof(float));
    float*    rinvA= (float*)   carve((size_t)B_ * M_ * sizeof(float));
    float*    akkA = (float*)   carve((size_t)B_ * M_ * sizeof(float));

    k_src    <<<(B_ * N_ + 255) / 256, 256, 0, stream>>>(ope, Ws, al, fsh, elw);
    k_dst    <<<(B_ * M_ + 255) / 256, 256, 0, stream>>>(ma, Wd, ar, fdw, erw, ekkw);
    k_pass1  <<<B_ * NC, 32, 0, stream>>>(adj, bidx, proc, elw, erw, we, ae, pmax, psum);
    k_combine<<<(B_ * M_ + 255) / 256, 256, 0, stream>>>(pmax, psum, ekkw, MxA, rinvA, akkA);
    k_pass2  <<<B_ * 8, 256, 0, stream>>>(adj, bidx, proc, elw, fsh, fdw, erw,
                                          MxA, rinvA, akkA, we, ae, out);
}
